// RankGAT_26044681683634
// MI455X (gfx1250) — compile-verified, matrix operand order checked by reading, not run on hardware
//
#include <hip/hip_runtime.h>
#include <hip/hip_bf16.h>

typedef __attribute__((ext_vector_type(16))) _Float16 v16h;
typedef __attribute__((ext_vector_type(8)))  float    v8f;

#define N_NODES 50000
#define N_EDGES 800000
#define N_GRAPHS 512
#define F_IN 9
#define UNITS 64

// ---------------------------------------------------------------------------
// utility
// ---------------------------------------------------------------------------
__global__ void fill_k(float* __restrict__ p, float v, int n) {
    int i = blockIdx.x * blockDim.x + threadIdx.x;
    if (i < n) p[i] = v;
}

__device__ __forceinline__ void atomicMaxFloat(float* addr, float val) {
    // monotonic-bit trick: works for mixed signs, init must be -inf
    if (val >= 0.0f) atomicMax((int*)addr, __float_as_int(val));
    else             atomicMin((unsigned int*)addr, __float_as_uint(val));
}

// branchless tanh: one v_exp_f32, no exec-mask divergence
__device__ __forceinline__ float fast_tanh(float x) {
    float e = __expf(2.0f * x);
    return 1.0f - 2.0f / (e + 1.0f);
}

// ---------------------------------------------------------------------------
// GAT input transform: h = x @ W_gat  [N,9]x[9,64]; alpha_s/d = h . a_src/dst
// one 64-thread block per node
// ---------------------------------------------------------------------------
__global__ void gat_in_k(const float* __restrict__ x, const float* __restrict__ Wg,
                         const float* __restrict__ aw_s, const float* __restrict__ aw_d,
                         float* __restrict__ h, float* __restrict__ alpha_s,
                         float* __restrict__ alpha_d) {
    int n = blockIdx.x;
    int u = threadIdx.x;  // 0..63
    __shared__ float xr[F_IN];
    __shared__ float ss[UNITS];
    __shared__ float sd[UNITS];
    if (u < F_IN) xr[u] = x[n * F_IN + u];
    __syncthreads();
    float acc = 0.f;
#pragma unroll
    for (int k = 0; k < F_IN; ++k) acc += xr[k] * Wg[k * UNITS + u];
    h[n * UNITS + u] = acc;
    ss[u] = acc * aw_s[u];
    sd[u] = acc * aw_d[u];
    __syncthreads();
    for (int s = 32; s > 0; s >>= 1) {
        if (u < s) { ss[u] += ss[u + s]; sd[u] += sd[u + s]; }
        __syncthreads();
    }
    if (u == 0) { alpha_s[n] = ss[0]; alpha_d[n] = sd[0]; }
}

// ---------------------------------------------------------------------------
// GAT edge kernels (edge id >= nE encodes self loop node = id - nE)
// ---------------------------------------------------------------------------
__global__ void gat_edge_max_k(const int* __restrict__ src, const int* __restrict__ dst,
                               const float* __restrict__ as, const float* __restrict__ ad,
                               float* __restrict__ e, float* __restrict__ mx,
                               int nE, int nN) {
    int i = blockIdx.x * blockDim.x + threadIdx.x;
    if (i >= nE + nN) return;
    int s = (i < nE) ? src[i] : (i - nE);
    int d = (i < nE) ? dst[i] : (i - nE);
    float v = as[s] + ad[d];
    v = (v > 0.f) ? v : 0.2f * v;        // leaky_relu 0.2
    e[i] = v;
    atomicMaxFloat(&mx[d], v);
}

__global__ void gat_edge_exp_k(const int* __restrict__ dst, const float* __restrict__ e,
                               const float* __restrict__ mx, float* __restrict__ ex,
                               float* __restrict__ denom, int nE, int nN) {
    int i = blockIdx.x * blockDim.x + threadIdx.x;
    if (i >= nE + nN) return;
    int d = (i < nE) ? dst[i] : (i - nE);
    float v = __expf(e[i] - mx[d]);
    ex[i] = v;
    atomicAdd(&denom[d], v);
}

// 16 threads per edge, float4 each: out[dst] += h[src] * coef
__global__ void gat_scatter_k(const int* __restrict__ src, const int* __restrict__ dst,
                              const float* __restrict__ ex, const float* __restrict__ denom,
                              const float* __restrict__ h, float* __restrict__ out,
                              int nE, int nN) {
    int gid = blockIdx.x * blockDim.x + threadIdx.x;
    int edge = gid >> 4;
    int u = (gid & 15) * 4;
    if (edge >= nE + nN) return;
    int s = (edge < nE) ? src[edge] : (edge - nE);
    int d = (edge < nE) ? dst[edge] : (edge - nE);
    float coef = ex[edge] / denom[d];
    const float4 hv = *(const float4*)&h[s * UNITS + u];
    atomicAdd(&out[d * UNITS + u + 0], hv.x * coef);
    atomicAdd(&out[d * UNITS + u + 1], hv.y * coef);
    atomicAdd(&out[d * UNITS + u + 2], hv.z * coef);
    atomicAdd(&out[d * UNITS + u + 3], hv.w * coef);
}

// ---------------------------------------------------------------------------
// GCN degree / scatter
// ---------------------------------------------------------------------------
__global__ void deg_accum_k(const int* __restrict__ dst, float* __restrict__ deg, int nE) {
    int i = blockIdx.x * blockDim.x + threadIdx.x;
    if (i < nE) atomicAdd(&deg[dst[i]], 1.0f);
}

__global__ void dinv_k(const float* __restrict__ deg, float* __restrict__ dinv, int n) {
    int i = blockIdx.x * blockDim.x + threadIdx.x;
    if (i < n) dinv[i] = (deg[i] > 0.f) ? rsqrtf(deg[i]) : 0.f;
}

__global__ void gcn_scatter_k(const int* __restrict__ src, const int* __restrict__ dst,
                              const float* __restrict__ dinv, const float* __restrict__ h,
                              float* __restrict__ out, int nE, int nN) {
    int gid = blockIdx.x * blockDim.x + threadIdx.x;
    int edge = gid >> 4;
    int u = (gid & 15) * 4;
    if (edge >= nE + nN) return;
    int s = (edge < nE) ? src[edge] : (edge - nE);
    int d = (edge < nE) ? dst[edge] : (edge - nE);
    float norm = dinv[s] * dinv[d];
    const float4 hv = *(const float4*)&h[s * UNITS + u];
    atomicAdd(&out[d * UNITS + u + 0], hv.x * norm);
    atomicAdd(&out[d * UNITS + u + 1], hv.y * norm);
    atomicAdd(&out[d * UNITS + u + 2], hv.z * norm);
    atomicAdd(&out[d * UNITS + u + 3], hv.w * norm);
}

__global__ void bias_act_k(const float* __restrict__ in, const float* __restrict__ b,
                           float* __restrict__ out, int total, int cols, int act) {
    int i = blockIdx.x * blockDim.x + threadIdx.x;
    if (i >= total) return;
    float v = in[i] + b[i % cols];
    if (act) v = fast_tanh(v);
    out[i] = v;
}

// ---------------------------------------------------------------------------
// Weight pre-pack: W[64, ncols] f32 row-major -> f16 fragments in the exact
// per-lane WMMA B layout. Record (t = col tile, c = k chunk, lane) holds the
// 16 halves lane `lane` feeds to v_wmma for that fragment.
// Launch: <<< nTiles, dim3(32, 2) >>>
// ---------------------------------------------------------------------------
__global__ void pack_w_k(const float* __restrict__ W, _Float16* __restrict__ P, int ncols) {
    int t = blockIdx.x;
    int c = threadIdx.y;       // k chunk
    int lane = threadIdx.x;    // 0..31
    int m = lane & 15;
    int kbase = (lane < 16) ? 0 : 8;
    int col = t * 16 + m;
    _Float16* rec = P + (((size_t)t * 2 + c) * 32 + lane) * 16;
#pragma unroll
    for (int j = 0; j < 16; ++j) {
        int kk = c * 32 + kbase + ((j < 8) ? j : (8 + j));
        rec[j] = (_Float16)W[kk * ncols + col];
    }
}

// ---------------------------------------------------------------------------
// WMMA GEMM:  out[nrows, NCOLS] = act(in[nrows, 64] @ W[64, NCOLS] + bias)
// One wave per 16-row tile; requires nrows % 16 == 0 (N=50000=3125*16).
// A: vectorized float4 loads + f16 convert. B: pre-packed f16 fragments.
// ---------------------------------------------------------------------------
template <int NCOLS, int ACT>
__global__ void gemm_wmma_k(const float* __restrict__ in, const _Float16* __restrict__ P,
                            const float* __restrict__ bias, float* __restrict__ out,
                            int nrows) {
    const int lane = threadIdx.x & 31;
    const int wave = threadIdx.x >> 5;
    const int rowTile = blockIdx.x * (blockDim.x >> 5) + wave;
    const int row0 = rowTile * 16;
    if (row0 >= nrows) return;   // full tiles only beyond this point

    const int m = lane & 15;
    const int kbase = (lane < 16) ? 0 : 8;
    const float* rp = in + (size_t)(row0 + m) * UNITS + kbase;

    // A fragment halves j: j<8 -> K=kbase+j (contiguous), j>=8 -> K=kbase+16+(j-8)
    union { float4 f4[4]; float f[16]; } ua0, ua1;
    ua0.f4[0] = *(const float4*)(rp + 0);
    ua0.f4[1] = *(const float4*)(rp + 4);
    ua0.f4[2] = *(const float4*)(rp + 16);
    ua0.f4[3] = *(const float4*)(rp + 20);
    ua1.f4[0] = *(const float4*)(rp + 32);
    ua1.f4[1] = *(const float4*)(rp + 36);
    ua1.f4[2] = *(const float4*)(rp + 48);
    ua1.f4[3] = *(const float4*)(rp + 52);
    v16h a0, a1;
#pragma unroll
    for (int j = 0; j < 16; ++j) { a0[j] = (_Float16)ua0.f[j]; a1[j] = (_Float16)ua1.f[j]; }

    constexpr int NT = NCOLS / 16;
#pragma unroll
    for (int t = 0; t < NT; ++t) {
        union { float4 f4[2]; v16h h; } ub0, ub1;
        const float4* r0 = (const float4*)(P + (((size_t)t * 2 + 0) * 32 + lane) * 16);
        const float4* r1 = (const float4*)(P + (((size_t)t * 2 + 1) * 32 + lane) * 16);
        ub0.f4[0] = r0[0]; ub0.f4[1] = r0[1];
        ub1.f4[0] = r1[0]; ub1.f4[1] = r1[1];

        v8f c = {};
        c = __builtin_amdgcn_wmma_f32_16x16x32_f16(false, a0, false, ub0.h, (short)0, c, false, false);
        c = __builtin_amdgcn_wmma_f32_16x16x32_f16(false, a1, false, ub1.h, (short)0, c, false, false);

        const float bv = bias ? bias[t * 16 + m] : 0.f;
        const int hi = (lane >= 16) ? 8 : 0;
#pragma unroll
        for (int r = 0; r < 8; ++r) {
            float v = c[r] + bv;
            if (ACT) v = fast_tanh(v);
            out[(size_t)(row0 + r + hi) * NCOLS + t * 16 + m] = v;
        }
    }
}

// ---------------------------------------------------------------------------
// MLP head:  out[n] = x[n,32] . W3 + b3
// ---------------------------------------------------------------------------
__global__ void mlp3_k(const float* __restrict__ x, const float* __restrict__ W3,
                       const float* __restrict__ b3, float* __restrict__ out, int n) {
    int i = blockIdx.x * blockDim.x + threadIdx.x;
    if (i >= n) return;
    float acc = b3[0];
    const float4* xv = (const float4*)&x[i * 32];
#pragma unroll
    for (int q = 0; q < 8; ++q) {
        float4 v = xv[q];
        acc += v.x * W3[q * 4 + 0] + v.y * W3[q * 4 + 1] +
               v.z * W3[q * 4 + 2] + v.w * W3[q * 4 + 3];
    }
    out[i] = acc;
}

// ---------------------------------------------------------------------------
// mean pool + final sigmoid
// ---------------------------------------------------------------------------
__global__ void pool_accum_k(const float* __restrict__ hn, const int* __restrict__ batch,
                             float* __restrict__ sums, float* __restrict__ cnts, int n) {
    int i = blockIdx.x * blockDim.x + threadIdx.x;
    if (i >= n) return;
    int g = batch[i];
    atomicAdd(&sums[g], hn[i]);
    atomicAdd(&cnts[g], 1.0f);
}

__global__ void final_sigmoid_k(const float* __restrict__ sa, const float* __restrict__ ca,
                                const float* __restrict__ sb, const float* __restrict__ cb,
                                float* __restrict__ out, int G) {
    int g = blockIdx.x * blockDim.x + threadIdx.x;
    if (g >= G) return;
    float ua = sa[g] / fmaxf(ca[g], 1.0f);
    float ub = sb[g] / fmaxf(cb[g], 1.0f);
    float z = ub - ua;
    out[g] = 1.0f / (1.0f + __expf(-z));
}

// ---------------------------------------------------------------------------
// host orchestration
// ---------------------------------------------------------------------------
static inline int cdiv(int a, int b) { return (a + b - 1) / b; }

struct BranchScratch {
    float *buf0, *buf1, *buf2;                     // [N,64] each
    float *alpha_s, *alpha_d, *mx, *denom, *dinv;  // [N]
    float *ex;                                     // [E+N]
    const _Float16 *Pgcn0, *Pgcn1, *P1, *P2;       // packed weights
};

static void run_branch(const float* x, const int* ei, float* hnode,
                       const float* W_gat, const float* a_src, const float* a_dst,
                       const float* b_gat, const float* b_gcn,
                       const float* b1, const float* b2,
                       const float* W3, const float* b3,
                       const BranchScratch& S, hipStream_t stream) {
    const int N = N_NODES, E = N_EDGES;
    const int* src = ei;          // edge_index row 0
    const int* dst = ei + E;      // edge_index row 1
    const int EN = E + N;

    // ---- GAT ----
    gat_in_k<<<N, UNITS, 0, stream>>>(x, W_gat, a_src, a_dst, S.buf0, S.alpha_s, S.alpha_d);
    fill_k<<<cdiv(N, 256), 256, 0, stream>>>(S.mx, -INFINITY, N);
    fill_k<<<cdiv(N, 256), 256, 0, stream>>>(S.denom, 0.f, N);
    gat_edge_max_k<<<cdiv(EN, 256), 256, 0, stream>>>(src, dst, S.alpha_s, S.alpha_d, S.ex, S.mx, E, N);
    gat_edge_exp_k<<<cdiv(EN, 256), 256, 0, stream>>>(dst, S.ex, S.mx, S.ex, S.denom, E, N);
    fill_k<<<cdiv(N * UNITS, 256), 256, 0, stream>>>(S.buf1, 0.f, N * UNITS);
    gat_scatter_k<<<cdiv(EN * 16, 256), 256, 0, stream>>>(src, dst, S.ex, S.denom, S.buf0, S.buf1, E, N);
    bias_act_k<<<cdiv(N * UNITS, 256), 256, 0, stream>>>(S.buf1, b_gat, S.buf0, N * UNITS, UNITS, 1);

    // ---- degrees (shared by both GCN layers) ----
    fill_k<<<cdiv(N, 256), 256, 0, stream>>>(S.denom, 1.0f, N);   // self loop
    deg_accum_k<<<cdiv(E, 256), 256, 0, stream>>>(dst, S.denom, E);
    dinv_k<<<cdiv(N, 256), 256, 0, stream>>>(S.denom, S.dinv, N);

    // ---- 2x GCN ----
    const int gemmGrid = cdiv(N, 16 * 8);   // 8 waves per 256-thread block
    for (int l = 0; l < 2; ++l) {
        const _Float16* P = (l == 0) ? S.Pgcn0 : S.Pgcn1;
        gemm_wmma_k<UNITS, 0><<<gemmGrid, 256, 0, stream>>>(S.buf0, P, nullptr, S.buf2, N);
        fill_k<<<cdiv(N * UNITS, 256), 256, 0, stream>>>(S.buf1, 0.f, N * UNITS);
        gcn_scatter_k<<<cdiv(EN * 16, 256), 256, 0, stream>>>(src, dst, S.dinv, S.buf2, S.buf1, E, N);
        bias_act_k<<<cdiv(N * UNITS, 256), 256, 0, stream>>>(S.buf1, b_gcn + l * UNITS, S.buf0,
                                                             N * UNITS, UNITS, 1);
    }

    // ---- MLP head ----
    gemm_wmma_k<UNITS, 1><<<gemmGrid, 256, 0, stream>>>(S.buf0, S.P1, b1, S.buf2, N);  // 64->64 tanh
    gemm_wmma_k<32, 1><<<gemmGrid, 256, 0, stream>>>(S.buf2, S.P2, b2, S.buf0, N);     // 64->32 tanh
    mlp3_k<<<cdiv(N, 256), 256, 0, stream>>>(S.buf0, W3, b3, hnode, N);                // 32->1
}

extern "C" void kernel_launch(void* const* d_in, const int* in_sizes, int n_in,
                              void* d_out, int out_size, void* d_ws, size_t ws_size,
                              hipStream_t stream) {
    (void)in_sizes; (void)n_in; (void)out_size; (void)ws_size;
    const float* x_a   = (const float*)d_in[0];
    const float* x_b   = (const float*)d_in[1];
    const int*   ei_a  = (const int*)d_in[2];
    const int*   ei_b  = (const int*)d_in[3];
    const int*   bat_a = (const int*)d_in[4];
    const int*   bat_b = (const int*)d_in[5];
    const float* W_gat = (const float*)d_in[6];
    const float* a_src = (const float*)d_in[7];
    const float* a_dst = (const float*)d_in[8];
    const float* b_gat = (const float*)d_in[9];
    const float* W_gcn = (const float*)d_in[10];
    const float* b_gcn = (const float*)d_in[11];
    const float* W1    = (const float*)d_in[12];
    const float* b1    = (const float*)d_in[13];
    const float* W2    = (const float*)d_in[14];
    const float* b2    = (const float*)d_in[15];
    const float* W3    = (const float*)d_in[16];
    const float* b3    = (const float*)d_in[17];
    float* out = (float*)d_out;

    // carve workspace (d_ws is 256B-aligned; packed f16 weights first)
    const int N = N_NODES, E = N_EDGES, G = N_GRAPHS;
    float* w = (float*)d_ws;
    _Float16* Pgcn0 = (_Float16*)w; w += 2048;   // 4 tiles * 2 chunks * 32 lanes * 16 halves
    _Float16* Pgcn1 = (_Float16*)w; w += 2048;
    _Float16* P1    = (_Float16*)w; w += 2048;
    _Float16* P2    = (_Float16*)w; w += 1024;   // 2 tiles (32 cols)

    BranchScratch S;
    S.buf0    = w; w += N * UNITS;
    S.buf1    = w; w += N * UNITS;
    S.buf2    = w; w += N * UNITS;
    S.alpha_s = w; w += N;
    S.alpha_d = w; w += N;
    S.mx      = w; w += N;
    S.denom   = w; w += N;
    S.dinv    = w; w += N;
    S.ex      = w; w += (E + N);
    float* ha     = w; w += N;
    float* hb     = w; w += N;
    float* sums_a = w; w += G;
    float* cnt_a  = w; w += G;
    float* sums_b = w; w += G;
    float* cnt_b  = w; w += G;
    S.Pgcn0 = Pgcn0; S.Pgcn1 = Pgcn1; S.P1 = P1; S.P2 = P2;

    // pack weights into WMMA fragment layout (once per call, shared by branches)
    pack_w_k<<<4, dim3(32, 2), 0, stream>>>(W_gcn,                 Pgcn0, UNITS);
    pack_w_k<<<4, dim3(32, 2), 0, stream>>>(W_gcn + UNITS * UNITS, Pgcn1, UNITS);
    pack_w_k<<<4, dim3(32, 2), 0, stream>>>(W1,                    P1,    UNITS);
    pack_w_k<<<2, dim3(32, 2), 0, stream>>>(W2,                    P2,    32);

    // branch A then branch B (scratch reused sequentially)
    run_branch(x_a, ei_a, ha, W_gat, a_src, a_dst, b_gat, b_gcn, b1, b2, W3, b3, S, stream);
    run_branch(x_b, ei_b, hb, W_gat, a_src, a_dst, b_gat, b_gcn, b1, b2, W3, b3, S, stream);

    // pooling + sigmoid(ub - ua)
    fill_k<<<cdiv(4 * G, 256), 256, 0, stream>>>(sums_a, 0.f, 4 * G);  // sums..cnt contiguous
    pool_accum_k<<<cdiv(N, 256), 256, 0, stream>>>(ha, bat_a, sums_a, cnt_a, N);
    pool_accum_k<<<cdiv(N, 256), 256, 0, stream>>>(hb, bat_b, sums_b, cnt_b, N);
    final_sigmoid_k<<<cdiv(G, 256), 256, 0, stream>>>(sums_a, cnt_a, sums_b, cnt_b, out, G);
}
